// InteractionBlockForSMPLX_74766790689505
// MI455X (gfx1250) — matrix operand order repaired and checked by reading, not compiled, not run on hardware
//
#include <hip/hip_runtime.h>
#include <math.h>

// ---------------------------------------------------------------------------
// CDNA5 / gfx1250 implementation. wave32. WMMA bf16 16x16x32 GEMM core.
// All GEMM A-operands are produced in bf16 by the producer kernels, and the
// weights are pre-packed into per-lane B-fragment layout (bf16). The GEMM hot
// loop is conversion-free: async global->LDS A-tile copy (ASYNCcnt path),
// ds_load_b128 A-frags, global_load_b128 B-frags, 2x v_wmma per K step.
// ---------------------------------------------------------------------------

typedef __attribute__((ext_vector_type(16))) __bf16 v16bf;
typedef __attribute__((ext_vector_type(8)))  float  v8f;
typedef __attribute__((ext_vector_type(4)))  int    i32x4;

#if defined(__HIP_DEVICE_COMPILE__) && defined(__has_builtin)
#if __has_builtin(__builtin_amdgcn_global_load_async_to_lds_b128) && \
    __has_builtin(__builtin_amdgcn_s_wait_asynccnt)
#define HAVE_ASYNC 1
#endif
#endif
#ifndef HAVE_ASYNC
#define HAVE_ASYNC 0
#endif

// async builtin signature (probe-confirmed via diagnostic):
// (int4 AS1* src, int4 AS3* dst, imm offset, imm cpol)
#define ASG_I4(p) ((__attribute__((address_space(1))) i32x4*)(void*)(p))
#define ASL_I4(p) ((__attribute__((address_space(3))) i32x4*)(p))

#define T_TOK 25
#define DM    768
#define NH_   12
#define B_    4
#define LQ1   1024
#define LC_   5376
#define HID_  192
#define EPS_  1e-6f

// workspace pool offsets (in floats), all 16-float aligned
#define OFF_A    0ull
#define OFF_LNQ  (OFF_A)                       // inj: LN(q) bf16 (4096x768)
#define OFF_QBUF (OFF_A + 3145728ull)          // inj: q contiguous (fp32)
#define OFF_QNEW (OFF_A + 6291456ull)          // inj: q + gamma*a (fp32)
#define OFF_B    16515072ull                   // inj: LN(c); ext: off/aw; ffn: h2
#define OFF_C    33030144ull                   // inj: value; ext: msda out
#define OFF_D    49545216ull                   // inj: off/aw; ext: LN(feat)
#define OFF_E    52690944ull                   // inj: msda out; ext: value; ffn: h1
#define OFF_F    56819712ull                   // packed bf16 weights (ushort)

static __device__ __forceinline__ unsigned short f2bf(float f) {
  unsigned int u = __float_as_uint(f);
  u += 0x7FFFu + ((u >> 16) & 1u);             // round-to-nearest-even
  return (unsigned short)(u >> 16);
}

// ---------------------------------------------------------------------------
// pack W[N,K] (fp32, row-major) into per-lane B-fragment layout:
//   out[((n>>4)*(K>>5) + (k>>5))*512 + lane*16 + s] = bf16(W[n,k]),
//   lane = 16*((k>>3)&1) + (n&15), s = (k&7) + 8*((k>>4)&1)
// B-frag (16x16x32 bf16): lane = N (+16 for k-half), halfword = K (table order).
// ---------------------------------------------------------------------------
__global__ __launch_bounds__(256) void pack_w_kernel(
    const float* __restrict__ W, unsigned short* __restrict__ out,
    int N, int K) {
  int i = blockIdx.x * 256 + threadIdx.x;
  if (i >= N * K) return;
  const int n = i / K, k = i % K;
  const int KC = K >> 5;
  out[(((size_t)(n >> 4) * KC + (k >> 5)) << 9) + (((k >> 3) & 1) << 8) + ((n & 15) << 4) + (k & 7) + (((k >> 4) & 1) << 3)] =
      f2bf(W[i]);
}

// ---------------------------------------------------------------------------
// LayerNorm: one block (256 threads) per row of 768; bf16 output (GEMM A-op)
// ---------------------------------------------------------------------------
__global__ __launch_bounds__(256) void ln_kernel(
    const float* __restrict__ in, const float* __restrict__ g,
    const float* __restrict__ b, unsigned short* __restrict__ out) {
  __shared__ float s1[256], s2[256];
  const int row = blockIdx.x, tid = threadIdx.x;
  const float* p = in + (size_t)row * DM;
  float x0 = p[tid], x1 = p[tid + 256], x2 = p[tid + 512];
  s1[tid] = x0 + x1 + x2;
  s2[tid] = x0 * x0 + x1 * x1 + x2 * x2;
  __syncthreads();
  for (int s = 128; s > 0; s >>= 1) {
    if (tid < s) { s1[tid] += s1[tid + s]; s2[tid] += s2[tid + s]; }
    __syncthreads();
  }
  const float mean = s1[0] * (1.0f / DM);
  const float var  = s2[0] * (1.0f / DM) - mean * mean;
  const float rstd = rsqrtf(var + EPS_);
  unsigned short* o = out + (size_t)row * DM;
  o[tid]       = f2bf((x0 - mean) * rstd * g[tid]       + b[tid]);
  o[tid + 256] = f2bf((x1 - mean) * rstd * g[tid + 256] + b[tid + 256]);
  o[tid + 512] = f2bf((x2 - mean) * rstd * g[tid + 512] + b[tid + 512]);
}

// ---------------------------------------------------------------------------
// GEMM: C[M,N] = res + scale[n] * (A[M,K] @ W[N,K]^T + bias[n])
// A is bf16 row-major; Wp = packed bf16 fragments; C/res/bias fp32.
// block tile 128x16, 8 waves (one 16x16 C tile each), K step 64 (2 WMMA/iter).
// M % 128 == 0, N % 16 == 0, K % 64 == 0 for all call sites.
// ---------------------------------------------------------------------------
#define BM 128
#define BN 16
#define BK 64

__global__ __launch_bounds__(256) void gemm_bias_kernel(
    const unsigned short* __restrict__ A, const unsigned short* __restrict__ Wp,
    const float* __restrict__ bias, const float* __restrict__ scale,
    const float* __restrict__ residual, float* __restrict__ C,
    int M, int N, int K) {
  __shared__ unsigned short As[BM][BK];   // [m][k] bf16, 16 KB
  const int tid  = threadIdx.x;
  const int wave = tid >> 5;
  const int lane = tid & 31;
  const int m0 = blockIdx.x * BM;
  const int n0 = blockIdx.y * BN;
  const int KC = K >> 5;                  // K chunks of 32
  const unsigned short* wp0 = Wp + ((size_t)(n0 >> 4) * KC << 9);

  v8f acc = {};

  const int mr = (wave << 4) + (lane & 15);   // A frag row in tile
  const int kb = (lane >> 4) << 3;            // A frag K sub-offset (0 or 8)

  for (int k0 = 0; k0 < K; k0 += BK) {
    // stage bf16 A tile (16 KB = 1024 x 16B chunks; 4 per thread)
#pragma unroll
    for (int it = 0; it < 4; ++it) {
      const int j  = tid + it * 256;          // 16-byte chunk index
      const int r  = j >> 3;                  // 8 chunks per 64-halfword row
      const int c8 = (j & 7) << 3;
#if HAVE_ASYNC
      __builtin_amdgcn_global_load_async_to_lds_b128(
          ASG_I4(&A[(size_t)(m0 + r) * K + (k0 + c8)]),
          ASL_I4(&As[r][c8]), 0, 0);
#else
      *(uint4*)&As[r][c8] = *(const uint4*)&A[(size_t)(m0 + r) * K + (k0 + c8)];
#endif
    }
#if HAVE_ASYNC
    __builtin_amdgcn_s_wait_asynccnt(0);
#endif
    __syncthreads();

    // B fragments straight from global (L2-resident packed weights)
    const unsigned short* wpc = wp0 + ((size_t)(k0 >> 5) << 9) + (lane << 4);
    union { uint4 q[2]; v16bf v; } af0, af1, bf0, bf1;
    bf0.q[0] = *(const uint4*)(wpc);
    bf0.q[1] = *(const uint4*)(wpc + 8);
    bf1.q[0] = *(const uint4*)(wpc + 512);
    bf1.q[1] = *(const uint4*)(wpc + 520);

    // A fragments for the two K chunks
    af0.q[0] = *(const uint4*)&As[mr][kb];
    af0.q[1] = *(const uint4*)&As[mr][kb + 16];
    af1.q[0] = *(const uint4*)&As[mr][kb + 32];
    af1.q[1] = *(const uint4*)&As[mr][kb + 48];

    acc = __builtin_amdgcn_wmma_f32_16x16x32_bf16(false, af0.v, false, bf0.v,
                                                  (short)0, acc, false, false);
    acc = __builtin_amdgcn_wmma_f32_16x16x32_bf16(false, af1.v, false, bf1.v,
                                                  (short)0, acc, false, false);
    __syncthreads();
  }

  // D layout: VGPR r holds M = r + 8*(lane>=16), N = lane & 15
  const int nn = n0 + (lane & 15);
  const float bi = bias  ? bias[nn]  : 0.0f;
  const float sc = scale ? scale[nn] : 1.0f;
  const int mbase = m0 + (wave << 4) + ((lane >> 4) << 3);
#pragma unroll
  for (int r = 0; r < 8; ++r) {
    const size_t idx = (size_t)(mbase + r) * N + nn;
    float v = (acc[r] + bi) * sc;
    if (residual) v += residual[idx];
    C[idx] = v;
  }
}

// ---------------------------------------------------------------------------
// per-head softmax over groups of G (= L*NP) values (fp32 in/out)
// ---------------------------------------------------------------------------
__global__ __launch_bounds__(256) void softmax_groups_kernel(
    float* __restrict__ aw, int total, int G) {
  int i = blockIdx.x * 256 + threadIdx.x;
  if (i >= total) return;
  float* p = aw + (size_t)(i / NH_) * (NH_ * G) + (size_t)(i % NH_) * G;
  float mx = -3.4e38f;
  for (int g = 0; g < G; ++g) mx = fmaxf(mx, p[g]);
  float s = 0.0f;
  float e[12];
  for (int g = 0; g < G; ++g) { e[g] = __expf(p[g] - mx); s += e[g]; }
  const float inv = 1.0f / s;
  for (int g = 0; g < G; ++g) p[g] = e[g] * inv;
}

// ---------------------------------------------------------------------------
// MSDA bilinear sampling. one block per (b,lq); 192 threads = 12 heads x 16,
// each thread accumulates 4 channels; bf16 output (feeds the Wo GEMM).
// ---------------------------------------------------------------------------
__global__ __launch_bounds__(192) void msda_kernel(
    const float* __restrict__ value, const float* __restrict__ off,
    const float* __restrict__ aw, const float* __restrict__ ref,
    unsigned short* __restrict__ out, int Lq, int Lv, int L,
    int H0l, int W0l, int S0, int H1l, int W1l, int S1,
    int H2l, int W2l, int S2) {
  const int blk = blockIdx.x;            // b*Lq + lq
  const int b   = blk / Lq;
  const int t   = threadIdx.x;
  const int h   = t >> 4;
  const int c4  = (t & 15) << 2;
  const float rx = ref[(size_t)blk * 2 + 0];
  const float ry = ref[(size_t)blk * 2 + 1];
  const float* offp = off + (size_t)blk * (NH_ * L * 8) + (size_t)h * (L * 8);
  const float* awp  = aw  + (size_t)blk * (NH_ * L * 4) + (size_t)h * (L * 4);

  float a0 = 0.f, a1 = 0.f, a2 = 0.f, a3 = 0.f;
  for (int l = 0; l < L; ++l) {
    const int Hl = (l == 0) ? H0l : ((l == 1) ? H1l : H2l);
    const int Wl = (l == 0) ? W0l : ((l == 1) ? W1l : W2l);
    const int st = (l == 0) ? S0  : ((l == 1) ? S1  : S2);
    const float* vbase = value + ((size_t)b * Lv + st) * (NH_ * 64) + h * 64 + c4;
    for (int p = 0; p < 4; ++p) {
      const float ox = offp[(l * 4 + p) * 2 + 0];
      const float oy = offp[(l * 4 + p) * 2 + 1];
      const float w  = awp[l * 4 + p];
      const float x  = (rx + ox / (float)Wl) * (float)Wl - 0.5f;
      const float y  = (ry + oy / (float)Hl) * (float)Hl - 0.5f;
      const float xf = floorf(x), yf = floorf(y);
      const float fx = x - xf, fy = y - yf;
      const int x0i = (int)xf, y0i = (int)yf;
#pragma unroll
      for (int dy = 0; dy < 2; ++dy) {
#pragma unroll
        for (int dx = 0; dx < 2; ++dx) {
          const int xi = x0i + dx, yi = y0i + dy;
          const float bw = (dx ? fx : 1.0f - fx) * (dy ? fy : 1.0f - fy);
          const bool valid = (xi >= 0) && (xi < Wl) && (yi >= 0) && (yi < Hl);
          const int xc = min(max(xi, 0), Wl - 1);
          const int yc = min(max(yi, 0), Hl - 1);
          const float4 g = *(const float4*)(vbase + (size_t)(yc * Wl + xc) * (NH_ * 64));
          const float wv = valid ? (w * bw) : 0.0f;
          a0 += wv * g.x; a1 += wv * g.y; a2 += wv * g.z; a3 += wv * g.w;
        }
      }
    }
  }
  uint2 pk;
  pk.x = (unsigned)f2bf(a0) | ((unsigned)f2bf(a1) << 16);
  pk.y = (unsigned)f2bf(a2) | ((unsigned)f2bf(a3) << 16);
  *(uint2*)(out + (size_t)blk * DM + h * 64 + c4) = pk;
}

// ---------------------------------------------------------------------------
// depthwise 3x3 conv (SAME) per scale segment + bias + exact GELU; bf16 out
// ---------------------------------------------------------------------------
__global__ __launch_bounds__(256) void dwconv_gelu_kernel(
    const float* __restrict__ hin, const float* __restrict__ wgt,
    const float* __restrict__ bias, unsigned short* __restrict__ hout,
    int total) {
  int i = blockIdx.x * 256 + threadIdx.x;
  if (i >= total) return;
  const int ch = i % HID_;
  const int n  = (i / HID_) % LC_;
  const int b  = i / (HID_ * LC_);
  int base, Hs, Ws;
  if (n < 4096)      { base = 0;    Hs = 64; Ws = 64; }
  else if (n < 5120) { base = 4096; Hs = 32; Ws = 32; }
  else               { base = 5120; Hs = 16; Ws = 16; }
  const int r = n - base, y = r / Ws, x = r % Ws;
  float s = 0.0f;
#pragma unroll
  for (int dy = -1; dy <= 1; ++dy) {
#pragma unroll
    for (int dx = -1; dx <= 1; ++dx) {
      const int yy = y + dy, xx = x + dx;
      if (yy >= 0 && yy < Hs && xx >= 0 && xx < Ws)
        s += hin[((size_t)b * LC_ + base + yy * Ws + xx) * HID_ + ch] *
             wgt[ch * 9 + (dy + 1) * 3 + (dx + 1)];
    }
  }
  s += bias[ch];
  s = 0.5f * s * (1.0f + erff(s * 0.70710678118654752f));   // exact GELU
  hout[i] = f2bf(s);
}

// gather q = x[:, T:, :] into a contiguous (B*1024, 768) buffer (fp32)
__global__ __launch_bounds__(256) void gather_q_kernel(
    const float* __restrict__ x, float* __restrict__ q, int total) {
  int i = blockIdx.x * 256 + threadIdx.x;
  if (i >= total) return;
  const int col = i % DM, row = i / DM;
  const int b = row / LQ1, lq = row % LQ1;
  q[i] = x[((size_t)b * (T_TOK + LQ1) + T_TOK + lq) * DM + col];
}

// x_out[:, :T] = x[:, :T]; x_out[:, T:] = q_new
__global__ __launch_bounds__(256) void assemble_x_kernel(
    const float* __restrict__ x, const float* __restrict__ qnew,
    float* __restrict__ xout, int total) {
  int i = blockIdx.x * 256 + threadIdx.x;
  if (i >= total) return;
  const int col = i % DM, row = i / DM;
  const int b = row / (T_TOK + LQ1), tk = row % (T_TOK + LQ1);
  xout[i] = (tk < T_TOK) ? x[i]
                         : qnew[((size_t)b * LQ1 + (tk - T_TOK)) * DM + col];
}

// ---------------------------------------------------------------------------
extern "C" void kernel_launch(void* const* d_in, const int* in_sizes, int n_in,
                              void* d_out, int out_size, void* d_ws, size_t ws_size,
                              hipStream_t stream) {
  (void)in_sizes; (void)n_in; (void)out_size; (void)ws_size;
  const float* x    = (const float*)d_in[0];
  const float* c    = (const float*)d_in[1];
  const float* ref1 = (const float*)d_in[2];
  const float* ref2 = (const float*)d_in[3];
  const float* inj_qn_g = (const float*)d_in[4];
  const float* inj_qn_b = (const float*)d_in[5];
  const float* inj_fn_g = (const float*)d_in[6];
  const float* inj_fn_b = (const float*)d_in[7];
  const float* inj_Wv   = (const float*)d_in[8];
  const float* inj_bv   = (const float*)d_in[9];
  const float* inj_Woff = (const float*)d_in[10];
  const float* inj_boff = (const float*)d_in[11];
  const float* inj_Waw  = (const float*)d_in[12];
  const float* inj_baw  = (const float*)d_in[13];
  const float* inj_Wo   = (const float*)d_in[14];
  const float* inj_bo   = (const float*)d_in[15];
  const float* ext_qn_g = (const float*)d_in[16];
  const float* ext_qn_b = (const float*)d_in[17];
  const float* ext_fn_g = (const float*)d_in[18];
  const float* ext_fn_b = (const float*)d_in[19];
  const float* ext_Wv   = (const float*)d_in[20];
  const float* ext_bv   = (const float*)d_in[21];
  const float* ext_Woff = (const float*)d_in[22];
  const float* ext_boff = (const float*)d_in[23];
  const float* ext_Waw  = (const float*)d_in[24];
  const float* ext_baw  = (const float*)d_in[25];
  const float* ext_Wo   = (const float*)d_in[26];
  const float* ext_bo   = (const float*)d_in[27];
  const float* gamma    = (const float*)d_in[28];
  const float* ffn_n_g  = (const float*)d_in[29];
  const float* ffn_n_b  = (const float*)d_in[30];
  const float* fc1W     = (const float*)d_in[31];
  const float* fc1b     = (const float*)d_in[32];
  const float* dwW      = (const float*)d_in[33];
  const float* dwb      = (const float*)d_in[34];
  const float* fc2W     = (const float*)d_in[35];
  const float* fc2b     = (const float*)d_in[36];

  float* ws = (float*)d_ws;
  const int M1 = B_ * LQ1;    // 4096
  const int M2 = B_ * LC_;    // 21504
  const int X_TOTAL = B_ * (T_TOK + LQ1) * DM;   // 3,222,528
  float* x_out  = (float*)d_out;
  float* c2_out = (float*)d_out + X_TOTAL;

  unsigned short* lnq = (unsigned short*)(ws + OFF_LNQ);   // bf16 GEMM A-ops
  float* qbuf  = ws + OFF_QBUF;
  float* qnew  = ws + OFF_QNEW;
  unsigned short* lnf_i = (unsigned short*)(ws + OFF_B);
  float* val_i = ws + OFF_C;
  float* off_i = ws + OFF_D;
  float* aw_i  = ws + OFF_D + 1179648ull;
  unsigned short* msa_i = (unsigned short*)(ws + OFF_E);

  // -------- pack all weights to bf16 fragment layout (once per launch) -----
  unsigned short* pw = (unsigned short*)(ws + OFF_F);
  unsigned short* p_iWv   = pw;                 // 589824
  unsigned short* p_iWoff = pw + 589824ull;     // 221184
  unsigned short* p_iWaw  = pw + 811008ull;     // 110592
  unsigned short* p_iWo   = pw + 921600ull;     // 589824
  unsigned short* p_eWv   = pw + 1511424ull;    // 589824
  unsigned short* p_eWoff = pw + 2101248ull;    // 73728
  unsigned short* p_eWaw  = pw + 2174976ull;    // 36864
  unsigned short* p_eWo   = pw + 2211840ull;    // 589824
  unsigned short* p_fc1   = pw + 2801664ull;    // 147456
  unsigned short* p_fc2   = pw + 2949120ull;    // 147456
  {
    struct { const float* w; unsigned short* o; int N, K; } jobs[10] = {
      {inj_Wv, p_iWv, DM, DM},   {inj_Woff, p_iWoff, 288, DM},
      {inj_Waw, p_iWaw, 144, DM},{inj_Wo, p_iWo, DM, DM},
      {ext_Wv, p_eWv, DM, DM},   {ext_Woff, p_eWoff, 96, DM},
      {ext_Waw, p_eWaw, 48, DM}, {ext_Wo, p_eWo, DM, DM},
      {fc1W, p_fc1, HID_, DM},   {fc2W, p_fc2, DM, HID_},
    };
    for (int j = 0; j < 10; ++j) {
      int tot = jobs[j].N * jobs[j].K;
      pack_w_kernel<<<(tot + 255) / 256, 256, 0, stream>>>(
          jobs[j].w, jobs[j].o, jobs[j].N, jobs[j].K);
    }
  }

  // ---------------- injection ----------------
  {
    int tot = M1 * DM;
    gather_q_kernel<<<(tot + 255) / 256, 256, 0, stream>>>(x, qbuf, tot);
    ln_kernel<<<M1, 256, 0, stream>>>(qbuf, inj_qn_g, inj_qn_b, lnq);
    ln_kernel<<<M2, 256, 0, stream>>>(c, inj_fn_g, inj_fn_b, lnf_i);
    gemm_bias_kernel<<<dim3(M2 / BM, DM / BN), 256, 0, stream>>>(
        lnf_i, p_iWv, inj_bv, nullptr, nullptr, val_i, M2, DM, DM);
    gemm_bias_kernel<<<dim3(M1 / BM, 288 / BN), 256, 0, stream>>>(
        lnq, p_iWoff, inj_boff, nullptr, nullptr, off_i, M1, 288, DM);
    gemm_bias_kernel<<<dim3(M1 / BM, 144 / BN), 256, 0, stream>>>(
        lnq, p_iWaw, inj_baw, nullptr, nullptr, aw_i, M1, 144, DM);
    int sm = M1 * NH_;
    softmax_groups_kernel<<<(sm + 255) / 256, 256, 0, stream>>>(aw_i, sm, 12);
    msda_kernel<<<M1, 192, 0, stream>>>(val_i, off_i, aw_i, ref1, msa_i,
        LQ1, LC_, 3, 64, 64, 0, 32, 32, 4096, 16, 16, 5120);
    gemm_bias_kernel<<<dim3(M1 / BM, DM / BN), 256, 0, stream>>>(
        msa_i, p_iWo, inj_bo, gamma, qbuf, qnew, M1, DM, DM);
    assemble_x_kernel<<<(X_TOTAL + 255) / 256, 256, 0, stream>>>(x, qnew, x_out, X_TOTAL);
  }

  // ---------------- extraction ----------------
  {
    unsigned short* lnf_e = (unsigned short*)(ws + OFF_D);  // LN(q_new) bf16
    unsigned short* lnq_e = (unsigned short*)(ws + OFF_A);  // LN(c) bf16
    float* val_e = ws + OFF_E;
    float* off_e = ws + OFF_B;
    float* aw_e  = ws + OFF_B + 2064384ull;
    unsigned short* msa_e = (unsigned short*)(ws + OFF_C);
    ln_kernel<<<M1, 256, 0, stream>>>(qnew, ext_fn_g, ext_fn_b, lnf_e);
    ln_kernel<<<M2, 256, 0, stream>>>(c, ext_qn_g, ext_qn_b, lnq_e);
    gemm_bias_kernel<<<dim3(M1 / BM, DM / BN), 256, 0, stream>>>(
        lnf_e, p_eWv, ext_bv, nullptr, nullptr, val_e, M1, DM, DM);
    gemm_bias_kernel<<<dim3(M2 / BM, 96 / BN), 256, 0, stream>>>(
        lnq_e, p_eWoff, ext_boff, nullptr, nullptr, off_e, M2, 96, DM);
    gemm_bias_kernel<<<dim3(M2 / BM, 48 / BN), 256, 0, stream>>>(
        lnq_e, p_eWaw, ext_baw, nullptr, nullptr, aw_e, M2, 48, DM);
    int sm = M2 * NH_;
    softmax_groups_kernel<<<(sm + 255) / 256, 256, 0, stream>>>(aw_e, sm, 4);
    msda_kernel<<<M2, 192, 0, stream>>>(val_e, off_e, aw_e, ref2, msa_e,
        LC_, LQ1, 1, 32, 32, 0, 32, 32, 0, 32, 32, 0);
    gemm_bias_kernel<<<dim3(M2 / BM, DM / BN), 256, 0, stream>>>(
        msa_e, p_eWo, ext_bo, nullptr, c, c2_out, M2, DM, DM);
  }

  // ---------------- conv FFN ----------------
  {
    unsigned short* lnc = (unsigned short*)(ws + OFF_A);
    float* h1 = ws + OFF_E;
    unsigned short* h2 = (unsigned short*)(ws + OFF_B);
    ln_kernel<<<M2, 256, 0, stream>>>(c2_out, ffn_n_g, ffn_n_b, lnc);
    gemm_bias_kernel<<<dim3(M2 / BM, HID_ / BN), 256, 0, stream>>>(
        lnc, p_fc1, fc1b, nullptr, nullptr, h1, M2, HID_, DM);
    int tot = B_ * LC_ * HID_;
    dwconv_gelu_kernel<<<(tot + 255) / 256, 256, 0, stream>>>(h1, dwW, dwb, h2, tot);
    gemm_bias_kernel<<<dim3(M2 / BM, DM / BN), 256, 0, stream>>>(
        h2, p_fc2, fc2b, nullptr, c2_out, c2_out, M2, DM, HID_);
  }
}
